// ConditionalRandomField_50062138802653
// MI455X (gfx1250) — matrix operand order repaired and checked by reading, not compiled, not run on hardware
//
#include <hip/hip_runtime.h>
#include <hip/hip_bf16.h>
#include <stddef.h>

// CRF forward log-likelihood, B=1024, S=512, T=50 (padded to 64).
// Denominator scan as per-step GEMM on WMMA: W = exp(alpha-m) * P,
// P[i][j] = exp(trans[i][j]-colmax[j]) precomputed in f16.
// 2 x v_wmma_f32_16x16x32_f16 per wave per step; 2 barriers per step;
// fully branchless epilogue; emit loads software-pipelined one step ahead.

#define Bn 1024
#define Sn 512
#define Tn 50
#define TP 64
#define AST (TP + 4)   // alpha row stride in floats (68*4B = 272B, 16B-aligned)

typedef __attribute__((ext_vector_type(16))) _Float16 v16h;
typedef __attribute__((ext_vector_type(8)))  float    v8f;

// ---------------- setup: colmax + P (f16), runs every call ----------------
__global__ __launch_bounds__(64) void crf_setup(const float* __restrict__ trans,
                                                float* __restrict__ c_arr,
                                                _Float16* __restrict__ P) {
    int j = threadIdx.x;           // 0..63
    float c = -1e30f;
    if (j < Tn) {
        for (int i = 0; i < Tn; ++i) c = fmaxf(c, trans[i * Tn + j]);
    }
    c_arr[j] = (j < Tn) ? c : 0.f;
    for (int i = 0; i < TP; ++i) {
        float v = (i < Tn && j < Tn) ? __expf(trans[i * Tn + j] - c) : 0.f;
        P[i * TP + j] = (_Float16)v;
    }
}

// ---------------- denominator: forward scan with WMMA ----------------
__global__ __launch_bounds__(128) void crf_scan(
    const float* __restrict__ logits,        // [B,S,T]
    const unsigned char* __restrict__ mask,  // [B,S] (bool)
    const float* __restrict__ c_arr,         // [64]
    const _Float16* __restrict__ P,          // [64,64]
    const float* __restrict__ start_t,       // [T]
    const float* __restrict__ end_t,         // [T]
    float* __restrict__ denom_out)           // [B]
{
    __shared__ float               s_alpha[16][AST];
    __shared__ __align__(16) float s_m[16];          // per-row maxes
    __shared__ unsigned char       s_maskT[Sn * 16]; // mask transposed: [s][b]

    const int t    = threadIdx.x;
    const int w    = t >> 5;       // wave id = N tile (0..3)
    const int lane = t & 31;
    const int hi   = lane >> 4;
    const int ln   = lane & 15;
    const int b0   = blockIdx.x * 16;
    const int jcol = w * 16 + ln;              // this lane's tag column (0..63)
    const int jc   = (jcol < Tn) ? jcol : (Tn - 1);  // clamped (always-valid loads)
    const bool vcol = (jcol < Tn);

    // B fragments (loop-invariant). 16-bit B 32x16 layout:
    // element e holds K = hi*8 + e (e<8), 16 + hi*8 + (e-8) (e>=8), N = lane%16.
    v16h Bf0, Bf1;
#pragma unroll
    for (int e = 0; e < 16; ++e) {
        int k = (e < 8) ? (hi * 8 + e) : (16 + hi * 8 + (e - 8));
        Bf0[e] = P[k * TP + jcol];
        Bf1[e] = P[(k + 32) * TP + jcol];
    }
    const float cj = c_arr[jcol];

    // stage mask tile into LDS, transposed: s_maskT[s*16 + b]
    for (int e = t; e < 16 * Sn; e += 128) {
        int b = e >> 9, s = e & (Sn - 1);
        s_maskT[s * 16 + b] = mask[(size_t)(b0 + b) * Sn + s];
    }

    // alpha0 = start + logits[:,0,:]
    for (int e = t; e < 16 * TP; e += 128) {
        int b = e >> 6, i = e & 63;
        float v = -1e30f;
        if (i < Tn) v = start_t[i] + logits[(size_t)(b0 + b) * Sn * Tn + i];
        s_alpha[b][i] = v;
    }

    // per-lane global row bases for emit loads (8 accumulator rows)
    size_t ebase[8];
#pragma unroll
    for (int r = 0; r < 8; ++r)
        ebase[r] = ((size_t)(b0 + hi * 8 + r) * Sn) * Tn + jc;

    // preload emit for s=1 (pipelined)
    float emit_c[8];
#pragma unroll
    for (int r = 0; r < 8; ++r) emit_c[r] = logits[ebase[r] + (size_t)1 * Tn];

    __syncthreads();

    // this lane's own alpha values (it is the sole writer of these 8 cells)
    float oldv[8];
#pragma unroll
    for (int r = 0; r < 8; ++r) oldv[r] = s_alpha[hi * 8 + r][jcol];

    for (int s = 1; s < Sn; ++s) {
        // ---- read phase: build A fragments from alpha (no cross-wave sharing) ----
        // lane covers row `ln`, k in {hi*8..+7, 16+hi*8..+7, 32+hi*8..+7, 48+hi*8..+7}
        float buf[32];
        {
            const float4* rp = (const float4*)&s_alpha[ln][hi * 8];
#pragma unroll
            for (int u = 0; u < 4; ++u) {
                float4 a = rp[u * 4 + 0];
                float4 b = rp[u * 4 + 1];
                buf[u * 8 + 0] = a.x; buf[u * 8 + 1] = a.y;
                buf[u * 8 + 2] = a.z; buf[u * 8 + 3] = a.w;
                buf[u * 8 + 4] = b.x; buf[u * 8 + 5] = b.y;
                buf[u * 8 + 6] = b.z; buf[u * 8 + 7] = b.w;
            }
        }
        float lmax = buf[0];
#pragma unroll
        for (int q = 1; q < 32; ++q) lmax = fmaxf(lmax, buf[q]);
        float rowmax = fmaxf(lmax, __shfl_xor(lmax, 16, 32));  // full 64-wide row max
        if (t < 16) s_m[t] = rowmax;  // wave 0 lanes 0..15 publish row maxes

        v16h A0, A1;
#pragma unroll
        for (int e = 0; e < 8; ++e) {
            A0[e]     = (_Float16)__expf(buf[e]      - rowmax);
            A0[e + 8] = (_Float16)__expf(buf[8 + e]  - rowmax);
            A1[e]     = (_Float16)__expf(buf[16 + e] - rowmax);
            A1[e + 8] = (_Float16)__expf(buf[24 + e] - rowmax);
        }

        // ---- issue next-step emit loads early (software pipeline) ----
        const int sN = (s + 1 < Sn) ? (s + 1) : (Sn - 1);
        float emit_n[8];
#pragma unroll
        for (int r = 0; r < 8; ++r) emit_n[r] = logits[ebase[r] + (size_t)sN * Tn];
        // prefetch s+2
        const int sP = (s + 2 < Sn) ? (s + 2) : (Sn - 1);
#pragma unroll
        for (int r = 0; r < 8; ++r)
            __builtin_prefetch(&logits[ebase[r] + (size_t)sP * Tn], 0, 0);

        v8f acc = {0.f, 0.f, 0.f, 0.f, 0.f, 0.f, 0.f, 0.f};
        acc = __builtin_amdgcn_wmma_f32_16x16x32_f16(false, A0, false, Bf0,
                                                     (short)0, acc, false, false);
        acc = __builtin_amdgcn_wmma_f32_16x16x32_f16(false, A1, false, Bf1,
                                                     (short)0, acc, false, false);

        __syncthreads();  // all alpha reads done; s_m published

        // ---- write phase: alpha_new = log(W) + m + cj + emit, branchless ----
        float4 mA = *(const float4*)&s_m[hi * 8];
        float4 mB = *(const float4*)&s_m[hi * 8 + 4];
        float mrs[8] = {mA.x, mA.y, mA.z, mA.w, mB.x, mB.y, mB.z, mB.w};
        uint2 mkp = *(const uint2*)&s_maskT[s * 16 + hi * 8];  // 8 mask bytes
#pragma unroll
        for (int r = 0; r < 8; ++r) {
            unsigned mby = ((r < 4) ? (mkp.x >> (8 * r)) : (mkp.y >> (8 * (r - 4)))) & 0xffu;
            float mkf = (mby != 0u) ? 1.f : 0.f;
            float W   = fmaxf(acc[r], 1e-30f);
            float val = __logf(W) + mrs[r] + cj + emit_c[r];
            float sel = fmaf(mkf, val - oldv[r], oldv[r]);   // branchless keep/update
            float nv  = vcol ? sel : -1e30f;
            s_alpha[hi * 8 + r][jcol] = nv;
            oldv[r] = nv;
        }
        __syncthreads();  // writes visible for next step

#pragma unroll
        for (int r = 0; r < 8; ++r) emit_c[r] = emit_n[r];
    }

    // denom[b] = logsumexp_j(alpha[b][j] + end[j])
    if (t < 16) {
        float m = -1e30f;
        for (int j = 0; j < Tn; ++j) m = fmaxf(m, s_alpha[t][j] + end_t[j]);
        float sum = 0.f;
        for (int j = 0; j < Tn; ++j) sum += __expf(s_alpha[t][j] + end_t[j] - m);
        denom_out[b0 + t] = __logf(sum) + m;
    }
}

// ---------------- numerator: tag-path score, one wave per batch ----------------
__global__ __launch_bounds__(256) void crf_num(
    const float* __restrict__ logits, const int* __restrict__ tags,
    const unsigned char* __restrict__ mask, const float* __restrict__ trans,
    const float* __restrict__ start_t, const float* __restrict__ end_t,
    float* __restrict__ score_out)
{
    int gw   = blockIdx.x * 8 + (threadIdx.x >> 5);  // batch
    int lane = threadIdx.x & 31;
    if (gw >= Bn) return;
    size_t bS = (size_t)gw * Sn;

    float sum = 0.f;
    int   cnt = 0;
    for (int s = lane; s < Sn; s += 32) {
        int   cur = tags[bS + s];
        float m_s = (float)mask[bS + s];
        cnt += (int)mask[bS + s];
        float contrib = 0.f;
        if (s < Sn - 1) {
            int   nxt = tags[bS + s + 1];
            float m_n = (float)mask[bS + s + 1];
            contrib = trans[cur * Tn + nxt] * m_n
                    + logits[(bS + s) * Tn + cur] * m_s;
        }
        sum += contrib;
    }
    for (int off = 16; off; off >>= 1) {
        sum += __shfl_xor(sum, off, 32);
        cnt += __shfl_xor(cnt, off, 32);
    }
    if (lane == 0) {
        int last = cnt - 1;
        int lt   = tags[bS + last];
        float total = sum + start_t[tags[bS]] + end_t[lt]
                    + logits[(bS + Sn - 1) * Tn + lt] * (float)mask[bS + Sn - 1];
        score_out[gw] = total;
    }
}

// ---------------- final: sum(score - denom) ----------------
__global__ __launch_bounds__(256) void crf_reduce(const float* __restrict__ score,
                                                  const float* __restrict__ denom,
                                                  float* __restrict__ out)
{
    __shared__ float red[8];
    int t = threadIdx.x;
    float v = 0.f;
    for (int i = t; i < Bn; i += 256) v += score[i] - denom[i];
    for (int off = 16; off; off >>= 1) v += __shfl_xor(v, off, 32);
    if ((t & 31) == 0) red[t >> 5] = v;
    __syncthreads();
    if (t == 0) {
        float s = 0.f;
        for (int i = 0; i < 8; ++i) s += red[i];
        out[0] = s;
    }
}

extern "C" void kernel_launch(void* const* d_in, const int* in_sizes, int n_in,
                              void* d_out, int out_size, void* d_ws, size_t ws_size,
                              hipStream_t stream) {
    const float*         logits  = (const float*)d_in[0];
    const int*           tags    = (const int*)d_in[1];
    const unsigned char* mask    = (const unsigned char*)d_in[2];  // jnp.bool_ = 1 byte
    const float*         trans   = (const float*)d_in[3];
    const float*         start_t = (const float*)d_in[4];
    const float*         end_t   = (const float*)d_in[5];

    char* w = (char*)d_ws;
    float*     c_arr = (float*)(w);                       // 64 floats   (256 B)
    _Float16*  P     = (_Float16*)(w + 256);              // 64*64 halfs (8192 B)
    float*     denom = (float*)(w + 256 + 8192);          // 1024 floats
    float*     score = (float*)(w + 256 + 8192 + 4096);   // 1024 floats

    crf_setup<<<1, 64, 0, stream>>>(trans, c_arr, P);
    crf_scan<<<Bn / 16, 128, 0, stream>>>(logits, mask, c_arr, P, start_t, end_t, denom);
    crf_num<<<Bn / 8, 256, 0, stream>>>(logits, tags, mask, trans, start_t, end_t, score);
    crf_reduce<<<1, 256, 0, stream>>>(score, denom, (float*)d_out);
}
